// GrangerAttention_55825984913883
// MI455X (gfx1250) — matrix operand order repaired and checked by reading, not compiled, not run on hardware
//
#include <hip/hip_runtime.h>
#include <stdint.h>

// ---------------------------------------------------------------------------
// GrangerAttention for MI455X (gfx1250, wave32, WMMA).
//  - edge_features (268 MB) is the roofline term: streamed ONCE from HBM,
//    shared across the 4 heads via LDS, double-buffered with
//    global_load_async_to_lds_b128 (ASYNCcnt) to overlap HBM latency with
//    the WMMA/softmax compute.
//  - all matmuls via v_wmma_f32_16x16x32_bf16 (fp32 accumulate).
//  - flash-attention streaming over j, per-row stats via 16-lane shfl_xor.
// ---------------------------------------------------------------------------

typedef __attribute__((ext_vector_type(16))) __bf16 bvec16;
typedef __attribute__((ext_vector_type(8)))  __bf16 bvec8;
typedef __attribute__((ext_vector_type(8)))  float  fvec8;

#define NSEQ 2048
#define BATCH 4
#define HEADS 4
#define DHEAD 64
#define DMODEL 256

__device__ __forceinline__ fvec8 wmma_bf16(bvec16 a, bvec16 b, fvec8 c) {
  return __builtin_amdgcn_wmma_f32_16x16x32_bf16(false, a, false, b, (short)0, c,
                                                 false, false);
}

__device__ __forceinline__ bvec16 pack16(const __bf16* p0, const __bf16* p1) {
  bvec8 lo = *(const bvec8*)p0;
  bvec8 hi = *(const bvec8*)p1;
  bvec16 r;
#pragma unroll
  for (int i = 0; i < 8; i++) { r[i] = lo[i]; r[i + 8] = hi[i]; }
  return r;
}

// A fragment (16x32 bf16, row-major source). lane<16: K {0-7,16-23};
// lane>=16: K {8-15,24-31}. Per-lane row = (lane & 15).
__device__ __forceinline__ bvec16 loadA(const __bf16* base, int ld, int k0, int lane) {
  int m = lane & 15, sel = lane >> 4;
  const __bf16* p = base + (size_t)m * ld + k0 + sel * 8;
  return pack16(p, p + 16);
}

// B fragment (32x16 bf16); source stored so column n is contiguous over K.
// lanes 0-15: K {0-7,16-23} of column (lane&15); lanes 16-31: K {8-15,24-31}.
__device__ __forceinline__ bvec16 loadB(const __bf16* base, int ld, int k0, int lane) {
  int c = lane & 15, sel = lane >> 4;
  const __bf16* p = base + (size_t)c * ld + k0 + sel * 8;
  return pack16(p, p + 16);
}

__device__ __forceinline__ float rowmax16(float v) {
#pragma unroll
  for (int m = 1; m < 16; m <<= 1) v = fmaxf(v, __shfl_xor(v, m, 16));
  return v;
}
__device__ __forceinline__ float rowsum16(float v) {
#pragma unroll
  for (int m = 1; m < 16; m <<= 1) v += __shfl_xor(v, m, 16);
  return v;
}

__device__ __forceinline__ void wait_async0() {
#if __has_builtin(__builtin_amdgcn_s_wait_asynccnt)
  __builtin_amdgcn_s_wait_asynccnt(0);
#else
  asm volatile("s_wait_asynccnt 0x0" ::: "memory");
#endif
}

// Async HBM -> LDS copy of one 16-byte packet per call (per lane).
__device__ __forceinline__ void async_copy_b128(void* lds_dst, const void* gsrc) {
  uint32_t lds = (uint32_t)(uintptr_t)lds_dst;  // low 32 bits = LDS offset
  asm volatile("global_load_async_to_lds_b128 %0, %1, off"
               :
               : "v"(lds), "v"(gsrc)
               : "memory");
}

// ---------------------------------------------------------------------------
__global__ __launch_bounds__(256) void cvt_bf16_kernel(const float* __restrict__ s,
                                                       __bf16* __restrict__ d, int n) {
  int i = blockIdx.x * 256 + threadIdx.x;
  if (i < n) d[i] = (__bf16)s[i];
}

// ---------------------------------------------------------------------------
// QKV projection: one wave computes a 16x16 tile of x @ W^T (K=256 -> 8 WMMAs).
// which=0: Q (scaled by 1/sqrt(Dh)=0.125, layout [b,h,n,d])
// which=1: K (layout [b,h,n,d])
// which=2: V (stored transposed: [b,h,d,n])
__global__ __launch_bounds__(32) void qkv_gemm_kernel(
    const __bf16* __restrict__ xb, const __bf16* __restrict__ wq,
    const __bf16* __restrict__ wk, const __bf16* __restrict__ wv,
    __bf16* __restrict__ qb, __bf16* __restrict__ kb, __bf16* __restrict__ vt) {
  int lane = threadIdx.x;
  int mt = blockIdx.x;           // 0..511  (8192/16 row tiles)
  int nt = blockIdx.y;           // 0..15   (256/16 col tiles)
  int which = blockIdx.z;        // 0=Q 1=K 2=V
  const __bf16* w = (which == 0) ? wq : (which == 1) ? wk : wv;
  const __bf16* abase = xb + (size_t)mt * 16 * DMODEL;
  const __bf16* bbase = w + (size_t)nt * 16 * DMODEL;  // B col c = W row c
  fvec8 acc = {};
#pragma unroll
  for (int kbk = 0; kbk < 8; kbk++) {
    bvec16 a = loadA(abase, DMODEL, kbk * 32, lane);
    bvec16 b = loadB(bbase, DMODEL, kbk * 32, lane);
    acc = wmma_bf16(a, b, acc);
  }
  int n = lane & 15, half = lane >> 4;
#pragma unroll
  for (int v = 0; v < 8; v++) {
    int m = half * 8 + v;
    int r = mt * 16 + m;              // global row in [0, B*N)
    int c = nt * 16 + n;              // global col in [0, 256)
    int b_ = r >> 11, nn = r & (NSEQ - 1);
    int h = c >> 6, d = c & 63;
    float val = acc[v];
    if (which == 0)
      qb[(((size_t)b_ * HEADS + h) * NSEQ + nn) * DHEAD + d] = (__bf16)(val * 0.125f);
    else if (which == 1)
      kb[(((size_t)b_ * HEADS + h) * NSEQ + nn) * DHEAD + d] = (__bf16)val;
    else
      vt[(((size_t)b_ * HEADS + h) * DHEAD + d) * NSEQ + nn] = (__bf16)val;
  }
}

// ---------------------------------------------------------------------------
// Fused edge-bias flash attention. Block = 4 waves (one per head) sharing one
// (b, i-tile). Edge tiles double-buffered in LDS via async global->LDS loads.
__global__ __launch_bounds__(128) void attn_kernel(
    const __bf16* __restrict__ qb, const __bf16* __restrict__ kb,
    const __bf16* __restrict__ vt, const float* __restrict__ edge,
    const float* __restrict__ We, const float* __restrict__ be,
    __bf16* __restrict__ ao) {
  __shared__ float4 eT[2][32][16];       // double-buffered edge tile   (16 KB)
  __shared__ __bf16 pT[HEADS][16][32];   // per-wave P staging C->A     ( 4 KB)

  int lane = threadIdx.x;
  int h = threadIdx.y;
  int i0 = blockIdx.x * 16;
  int b = blockIdx.y;
  int n = lane & 15, half = lane >> 4;
  int t = h * 32 + lane;  // 0..127

  float we0 = We[h * 4 + 0], we1 = We[h * 4 + 1];
  float we2 = We[h * 4 + 2], we3 = We[h * 4 + 3];
  float beh = be[h];

  size_t bh = (size_t)b * HEADS + h;
  const __bf16* Qb = qb + (bh * NSEQ + i0) * DHEAD;
  const __bf16* Kb = kb + bh * NSEQ * DHEAD;
  const __bf16* Vb = vt + bh * DHEAD * NSEQ;

  bvec16 aq0 = loadA(Qb, DHEAD, 0, lane);
  bvec16 aq1 = loadA(Qb, DHEAD, 32, lane);

  fvec8 oAcc[4];
#pragma unroll
  for (int dt = 0; dt < 4; dt++) oAcc[dt] = fvec8{};
  float mrow[8], lrow[8];
#pragma unroll
  for (int v = 0; v < 8; v++) { mrow[v] = -1e30f; lrow[v] = 0.0f; }

  // Prefetch tile 0 asynchronously into buffer 0: 32 j-rows x 16 i x float4.
#pragma unroll
  for (int q = t; q < 512; q += 128) {
    int jj = q >> 4, ii = q & 15;
    async_copy_b128(&eT[0][jj][ii],
                    edge + (((size_t)b * NSEQ + jj) * NSEQ + (i0 + ii)) * 4);
  }

  for (int jt = 0; jt < NSEQ / 32; jt++) {
    int j0 = jt * 32;
    int buf = jt & 1;
    wait_async0();       // this wave's async tile loads have landed in LDS
    __syncthreads();     // all waves' tiles landed; prior buffer fully read

    // Kick off async prefetch of the NEXT edge tile into the other buffer.
    if (jt + 1 < NSEQ / 32) {
      int j0n = j0 + 32;
#pragma unroll
      for (int q = t; q < 512; q += 128) {
        int jj = q >> 4, ii = q & 15;
        async_copy_b128(&eT[buf ^ 1][jj][ii],
                        edge + (((size_t)b * NSEQ + (j0n + jj)) * NSEQ + (i0 + ii)) * 4);
      }
    }

    // S = (Q*scale) @ K^T over K=64 (2 WMMAs per 16-col tile).
    fvec8 s0 = {}, s1 = {};
    {
      bvec16 bk = loadB(Kb + (size_t)j0 * DHEAD, DHEAD, 0, lane);
      s0 = wmma_bf16(aq0, bk, s0);
      bk = loadB(Kb + (size_t)j0 * DHEAD, DHEAD, 32, lane);
      s0 = wmma_bf16(aq1, bk, s0);
      bk = loadB(Kb + (size_t)(j0 + 16) * DHEAD, DHEAD, 0, lane);
      s1 = wmma_bf16(aq0, bk, s1);
      bk = loadB(Kb + (size_t)(j0 + 16) * DHEAD, DHEAD, 32, lane);
      s1 = wmma_bf16(aq1, bk, s1);
    }

    // Edge bias: bias[i,j] = edge[b,j,i,:] . We[h] + be[h]   (note j,i swap).
#pragma unroll
    for (int v = 0; v < 8; v++) {
      int m = half * 8 + v;
      float4 e0 = eT[buf][n][m];
      float4 e1 = eT[buf][n + 16][m];
      s0[v] += e0.x * we0 + e0.y * we1 + e0.z * we2 + e0.w * we3 + beh;
      s1[v] += e1.x * we0 + e1.y * we1 + e1.z * we2 + e1.w * we3 + beh;
    }

    // Online softmax (stats per row m = half*8+v, aligned with C layout).
#pragma unroll
    for (int v = 0; v < 8; v++) {
      float tmax = rowmax16(fmaxf(s0[v], s1[v]));
      float mnew = fmaxf(mrow[v], tmax);
      float alpha = __expf(mrow[v] - mnew);
      mrow[v] = mnew;
      float p0 = __expf(s0[v] - mnew);
      float p1 = __expf(s1[v] - mnew);
      lrow[v] = lrow[v] * alpha + rowsum16(p0 + p1);
#pragma unroll
      for (int dt = 0; dt < 4; dt++) oAcc[dt][v] *= alpha;
      int m = half * 8 + v;
      pT[h][m][n] = (__bf16)p0;        // C-format -> row-major LDS
      pT[h][m][n + 16] = (__bf16)p1;
    }

    // Re-read P in A format (per-wave LDS region; LDS in-order within wave).
    bvec16 ap = loadA(&pT[h][0][0], 32, 0, lane);

    // O += P @ V  (V transposed => contiguous B-fragment loads), 4 d-tiles.
#pragma unroll
    for (int dt = 0; dt < 4; dt++) {
      bvec16 bv = loadB(Vb + (size_t)dt * 16 * NSEQ, NSEQ, j0, lane);
      oAcc[dt] = wmma_bf16(ap, bv, oAcc[dt]);
    }
  }

  // Normalize, store bf16 attention output as [B*N, 256].
#pragma unroll
  for (int v = 0; v < 8; v++) {
    float inv = 1.0f / lrow[v];
    int m = half * 8 + v;
    size_t rowoff = ((size_t)b * NSEQ + i0 + m) * DMODEL + h * DHEAD;
#pragma unroll
    for (int dt = 0; dt < 4; dt++)
      ao[rowoff + dt * 16 + n] = (__bf16)(oAcc[dt][v] * inv);
  }
}

// ---------------------------------------------------------------------------
// Output projection: out = AO @ Wo^T + bo  -> fp32 d_out.
__global__ __launch_bounds__(32) void out_gemm_kernel(
    const __bf16* __restrict__ aob, const __bf16* __restrict__ wob,
    const float* __restrict__ bo, float* __restrict__ out) {
  int lane = threadIdx.x;
  int mt = blockIdx.x;  // 0..511
  int nt = blockIdx.y;  // 0..15
  const __bf16* abase = aob + (size_t)mt * 16 * DMODEL;
  const __bf16* bbase = wob + (size_t)nt * 16 * DMODEL;
  fvec8 acc = {};
#pragma unroll
  for (int kbk = 0; kbk < 8; kbk++) {
    bvec16 a = loadA(abase, DMODEL, kbk * 32, lane);
    bvec16 b = loadB(bbase, DMODEL, kbk * 32, lane);
    acc = wmma_bf16(a, b, acc);
  }
  int n = lane & 15, half = lane >> 4;
#pragma unroll
  for (int v = 0; v < 8; v++) {
    int m = half * 8 + v;
    int r = mt * 16 + m;
    int c = nt * 16 + n;
    out[(size_t)r * DMODEL + c] = acc[v] + bo[c];
  }
}

// ---------------------------------------------------------------------------
extern "C" void kernel_launch(void* const* d_in, const int* in_sizes, int n_in,
                              void* d_out, int out_size, void* d_ws, size_t ws_size,
                              hipStream_t stream) {
  (void)in_sizes; (void)n_in; (void)out_size; (void)ws_size;
  const float* x    = (const float*)d_in[0];
  const float* edge = (const float*)d_in[1];
  const float* Wq   = (const float*)d_in[2];
  const float* Wk   = (const float*)d_in[3];
  const float* Wv   = (const float*)d_in[4];
  const float* Wo   = (const float*)d_in[5];
  const float* bo   = (const float*)d_in[6];
  const float* We   = (const float*)d_in[7];
  const float* be   = (const float*)d_in[8];
  float* out = (float*)d_out;

  const size_t ROWS = (size_t)BATCH * NSEQ;        // 8192
  size_t off = 0;
  auto take = [&](size_t bytes) -> char* {
    char* p = (char*)d_ws + off;
    off += (bytes + 255) & ~(size_t)255;
    return p;
  };
  __bf16* xb  = (__bf16*)take(ROWS * DMODEL * 2);              // 4 MB
  __bf16* wqb = (__bf16*)take(DMODEL * DMODEL * 2);
  __bf16* wkb = (__bf16*)take(DMODEL * DMODEL * 2);
  __bf16* wvb = (__bf16*)take(DMODEL * DMODEL * 2);
  __bf16* wob = (__bf16*)take(DMODEL * DMODEL * 2);
  __bf16* qbf = (__bf16*)take(ROWS * DMODEL * 2);              // [b,h,n,d]
  __bf16* kbf = (__bf16*)take(ROWS * DMODEL * 2);              // [b,h,n,d]
  __bf16* vtb = (__bf16*)take(ROWS * DMODEL * 2);              // [b,h,d,n]
  __bf16* aob = (__bf16*)take(ROWS * DMODEL * 2);              // [B*N, 256]

  // fp32 -> bf16 converts
  {
    int nx = (int)(ROWS * DMODEL);
    cvt_bf16_kernel<<<(nx + 255) / 256, 256, 0, stream>>>(x, xb, nx);
    int nw = DMODEL * DMODEL;
    cvt_bf16_kernel<<<(nw + 255) / 256, 256, 0, stream>>>(Wq, wqb, nw);
    cvt_bf16_kernel<<<(nw + 255) / 256, 256, 0, stream>>>(Wk, wkb, nw);
    cvt_bf16_kernel<<<(nw + 255) / 256, 256, 0, stream>>>(Wv, wvb, nw);
    cvt_bf16_kernel<<<(nw + 255) / 256, 256, 0, stream>>>(Wo, wob, nw);
  }

  // QKV projection (Q scaled by 1/8; V stored transposed)
  qkv_gemm_kernel<<<dim3(ROWS / 16, DMODEL / 16, 3), 32, 0, stream>>>(
      xb, wqb, wkb, wvb, qbf, kbf, vtb);

  // Fused edge-bias flash attention (async double-buffered edge stream)
  attn_kernel<<<dim3(NSEQ / 16, BATCH), dim3(32, HEADS), 0, stream>>>(
      qbf, kbf, vtb, edge, We, be, aob);

  // Output projection + bias -> fp32
  out_gemm_kernel<<<dim3(ROWS / 16, DMODEL / 16), 32, 0, stream>>>(aob, wob, bo, out);
}
